// Molformer_7421703488130
// MI455X (gfx1250) — matrix operand order repaired and checked by reading, not compiled, not run on hardware
//
#include <hip/hip_runtime.h>
#include <hip/hip_bf16.h>

// ---------------- model dims ----------------
#define BB 64
#define TT 512
#define VV 2362
#define DD 768
#define HH 12
#define LL 12
#define FF 32
#define DH 64
#define MM (BB * TT)   // 32768 rows

// ---------------- WMMA types ----------------
typedef __attribute__((ext_vector_type(16))) __bf16 v16bf;
typedef __attribute__((ext_vector_type(8)))  __bf16 v8bf;
typedef __attribute__((ext_vector_type(8)))  float  v8f;

// ---------------- GEMM: C[M,N] = A[M,K] @ W[K,N] + bias, bf16 WMMA, fp32 acc ----
// block = 256 threads = 8 waves; block tile 128x128; wave tile 64x32 (4x2 WMMA tiles)
// register-staged + LDS double-buffered pipeline:
//   iter k: issue global loads for tile k+1 into regs
//           -> ds_load fragments(buf k) + 8x WMMA   (global loads in flight)
//           -> cvt+ds_store regs into buf k^1       (loadcnt wait lands here)
//           -> barrier
#define TM 128
#define TN 128
#define TK 32
#define LSTR 40   // bf16 LDS row stride (pad kills bank conflicts, keeps 16B align)

template <int MODE /*0=none,1=gelu*/>
__global__ __launch_bounds__(256) void k_gemm(const float* __restrict__ A,
                                              const float* __restrict__ W,
                                              const float* __restrict__ bias,
                                              float* __restrict__ C,
                                              int Mdim, int Ndim, int Kdim) {
  __shared__ __bf16 As[2][TM][LSTR];   // As[buf][row][k]
  __shared__ __bf16 Bs[2][TN][LSTR];   // Bs[buf][n][k]  (transposed stage)

  const int tid  = threadIdx.x;
  const int lane = tid & 31;
  const int wave = tid >> 5;
  const int waveM = wave >> 2;   // 0..1
  const int waveN = wave & 3;    // 0..3
  const int rowBase = blockIdx.y * TM;
  const int colBase = blockIdx.x * TN;
  (void)Mdim;

  v8f acc[4][2];
#pragma unroll
  for (int mi = 0; mi < 4; ++mi)
#pragma unroll
    for (int ni = 0; ni < 2; ++ni)
#pragma unroll
      for (int e = 0; e < 8; ++e) acc[mi][ni][e] = 0.f;

  // per-lane fragment geometry (ISA 7.12.2 layouts)
  const int arow  = lane & 15;
  const int abase = (lane >> 4) << 3;   // A: lanes 0-15 -> K {0..7,16..23}, lanes 16-31 -> K {8..15,24..31}
  const int brow  = lane & 15;
  const int bbase = (lane >> 4) << 3;   // B: lanes 0-15 -> K {0..7,16..23}, lanes 16-31 -> K {8..15,24..31}

  auto loadTile = [&](int k0, float4* ar, float4* br) {
#pragma unroll
    for (int i = 0; i < 4; ++i) {
      const int li = tid + i * 256;
      ar[i] = *(const float4*)(A + (size_t)(rowBase + (li >> 3)) * Kdim + k0 + ((li & 7) << 2));
      br[i] = *(const float4*)(W + (size_t)(k0 + (li >> 5)) * Ndim + colBase + ((li & 31) << 2));
    }
  };
  auto storeTile = [&](int buf, const float4* ar, const float4* br) {
#pragma unroll
    for (int i = 0; i < 4; ++i) {
      const int li = tid + i * 256;
      const int r  = li >> 3;
      const int c  = (li & 7) << 2;
      As[buf][r][c + 0] = (__bf16)ar[i].x; As[buf][r][c + 1] = (__bf16)ar[i].y;
      As[buf][r][c + 2] = (__bf16)ar[i].z; As[buf][r][c + 3] = (__bf16)ar[i].w;
      const int rb = li >> 5;            // k
      const int cb = (li & 31) << 2;     // n
      Bs[buf][cb + 0][rb] = (__bf16)br[i].x; Bs[buf][cb + 1][rb] = (__bf16)br[i].y;
      Bs[buf][cb + 2][rb] = (__bf16)br[i].z; Bs[buf][cb + 3][rb] = (__bf16)br[i].w;
    }
  };

  // prologue: stage tile 0
  {
    float4 ar[4], br[4];
    loadTile(0, ar, br);
    storeTile(0, ar, br);
  }
  __syncthreads();

  const int nk = Kdim / TK;
  for (int kt = 0; kt < nk; ++kt) {
    const int buf = kt & 1;
    const bool hasNext = (kt + 1 < nk);

    // issue global loads for next tile first (stay in flight across the WMMAs)
    float4 ar[4], br[4];
    if (hasNext) loadTile((kt + 1) * TK, ar, br);
    // prefetch 2 tiles ahead into near caches
    if (kt + 2 < nk) {
      __builtin_prefetch(A + (size_t)(rowBase + (tid >> 1)) * Kdim + (kt + 2) * TK, 0, 3);
      __builtin_prefetch(W + (size_t)((kt + 2) * TK + (tid & 31)) * Ndim + colBase, 0, 3);
    }

    // fragments from current buffer (deps: barrier + ds only)
    v16bf afrag[4], bfrag[2];
#pragma unroll
    for (int mi = 0; mi < 4; ++mi) {
      const __bf16* p = &As[buf][waveM * 64 + mi * 16 + arow][0];
      v8bf lo = *(const v8bf*)(p + abase);
      v8bf hi = *(const v8bf*)(p + abase + 16);
#pragma unroll
      for (int e = 0; e < 8; ++e) { afrag[mi][e] = lo[e]; afrag[mi][8 + e] = hi[e]; }
    }
#pragma unroll
    for (int ni = 0; ni < 2; ++ni) {
      const __bf16* p = &Bs[buf][waveN * 32 + ni * 16 + brow][0];
      v8bf lo = *(const v8bf*)(p + bbase);
      v8bf hi = *(const v8bf*)(p + bbase + 16);
#pragma unroll
      for (int e = 0; e < 8; ++e) { bfrag[ni][e] = lo[e]; bfrag[ni][8 + e] = hi[e]; }
    }

#pragma unroll
    for (int mi = 0; mi < 4; ++mi)
#pragma unroll
      for (int ni = 0; ni < 2; ++ni)
        acc[mi][ni] = __builtin_amdgcn_wmma_f32_16x16x32_bf16(
            false, afrag[mi], false, bfrag[ni], (short)0, acc[mi][ni], false, false);

    // now wait for the in-flight globals and commit them to the other buffer
    if (hasNext) storeTile(buf ^ 1, ar, br);

    __syncthreads();   // reads of buf done; writes of buf^1 visible
  }

  // epilogue; C f32 16x16 layout: lanes 0-15 N=lane M=i, lanes 16-31 N=lane-16 M=8+i
  const int cn = lane & 15;
  const int cm = (lane >> 4) << 3;
#pragma unroll
  for (int mi = 0; mi < 4; ++mi) {
#pragma unroll
    for (int ni = 0; ni < 2; ++ni) {
      const int col = colBase + waveN * 32 + ni * 16 + cn;
      const float bv = bias[col];
#pragma unroll
      for (int i = 0; i < 8; ++i) {
        const int row = rowBase + waveM * 64 + mi * 16 + cm + i;
        float v = acc[mi][ni][i] + bv;
        if (MODE == 1) v = 0.5f * v * (1.f + erff(v * 0.70710678118654752f));
        C[(size_t)row * Ndim + col] = v;
      }
    }
  }
}

// ---------------- embedding gather ----------------
__global__ __launch_bounds__(256) void k_embed(const int* __restrict__ ids,
                                               const float* __restrict__ emb,
                                               float* __restrict__ x) {
  const int bt = blockIdx.x;
  const int id = ids[bt];
  const float* src = emb + (size_t)id * DD;
  float* dst = x + (size_t)bt * DD;
  for (int j = threadIdx.x; j < DD; j += 256) dst[j] = src[j];
}

// ---------------- RoPE + ReLU feature map: qf/kf = relu(rope(q/k) @ omega) ----
__global__ __launch_bounds__(256) void k_rope_feat(const float* __restrict__ q,
                                                   const float* __restrict__ kb,
                                                   const float* __restrict__ om, // [DH][FF]
                                                   const int* __restrict__ mask,
                                                   float* __restrict__ qf,
                                                   float* __restrict__ kf) {
  __shared__ float qr[DD], kr[DD], lom[DH * FF];
  const int bt = blockIdx.x;
  const int t  = bt % TT;
  for (int j = threadIdx.x; j < DH * FF; j += 256) lom[j] = om[j];
  for (int j = threadIdx.x; j < DD; j += 256) {
    const int d = j & 63;
    const int i = d & 31;
    const float ang = (float)t * __expf(-0.28782313662425572f * (float)i); // 10000^(-i/32)
    const float c = cosf(ang), s = sinf(ang);
    const int pj = (d < 32) ? (j + 32) : (j - 32);
    const float qv = q[(size_t)bt * DD + j];
    const float qp = q[(size_t)bt * DD + pj];
    const float kv = kb[(size_t)bt * DD + j];
    const float kp = kb[(size_t)bt * DD + pj];
    const float rq = (d < 32) ? -qp : qp;
    const float rk = (d < 32) ? -kp : kp;
    qr[j] = qv * c + rq * s;
    kr[j] = kv * c + rk * s;
  }
  __syncthreads();
  const float mv = (float)mask[bt];
  for (int o = threadIdx.x; o < HH * FF; o += 256) {
    const int h = o >> 5, f = o & 31;
    float aq = 0.f, ak = 0.f;
#pragma unroll 8
    for (int d = 0; d < DH; ++d) {
      const float w = lom[d * FF + f];
      aq += qr[h * DH + d] * w;
      ak += kr[h * DH + d] * w;
    }
    qf[(size_t)bt * (HH * FF) + o] = fmaxf(aq, 0.f);
    kf[(size_t)bt * (HH * FF) + o] = fmaxf(ak, 0.f) * mv;
  }
}

// ---------------- kv[b,h,f,d] = sum_s kf*v ; ksum[b,h,f] = sum_s kf ----------
__global__ __launch_bounds__(256) void k_kv_reduce(const float* __restrict__ kf,
                                                   const float* __restrict__ v,
                                                   float* __restrict__ kv,
                                                   float* __restrict__ ksum) {
  __shared__ float Lkf[8][FF], Lv[8][DH];
  const int bh = blockIdx.x;
  const int b = bh / HH, h = bh % HH;
  const int tid = threadIdx.x;
  const int fi = tid >> 6;   // 0..3
  const int d  = tid & 63;
  float acc[8];
#pragma unroll
  for (int i = 0; i < 8; ++i) acc[i] = 0.f;
  float ks = 0.f;

  for (int s0 = 0; s0 < TT; s0 += 8) {
    Lkf[tid >> 5][tid & 31] =
        kf[((size_t)(b * TT + s0 + (tid >> 5)) * HH + h) * FF + (tid & 31)];
#pragma unroll
    for (int r = 0; r < 2; ++r) {
      const int li = tid + r * 256;
      Lv[li >> 6][li & 63] =
          v[((size_t)(b * TT + s0 + (li >> 6)) * HH + h) * DH + (li & 63)];
    }
    __syncthreads();
#pragma unroll
    for (int ss = 0; ss < 8; ++ss) {
      const float vd = Lv[ss][d];
#pragma unroll
      for (int i = 0; i < 8; ++i) acc[i] += Lkf[ss][fi + i * 4] * vd;
    }
    if (tid < FF) {
#pragma unroll
      for (int ss = 0; ss < 8; ++ss) ks += Lkf[ss][tid];
    }
    __syncthreads();
  }
#pragma unroll
  for (int i = 0; i < 8; ++i)
    kv[((size_t)bh * FF + (fi + i * 4)) * DH + d] = acc[i];
  if (tid < FF) ksum[(size_t)bh * FF + tid] = ks;
}

// ---------------- a[b,t,h,d] = (qf . kv[:,d]) / (qf . ksum + 1e-6) ----------
__global__ __launch_bounds__(256) void k_attn_out(const float* __restrict__ qf,
                                                  const float* __restrict__ kv,
                                                  const float* __restrict__ ksum,
                                                  float* __restrict__ a) {
  __shared__ float Lkv[FF * DH], Lks[FF];
  const int bh = blockIdx.x;
  const int b = bh / HH, h = bh % HH;
  const int tid = threadIdx.x;
  for (int j = tid; j < FF * DH; j += 256) Lkv[j] = kv[(size_t)bh * FF * DH + j];
  if (tid < FF) Lks[tid] = ksum[(size_t)bh * FF + tid];
  __syncthreads();
  const int tl = tid >> 6;   // 0..3
  const int d  = tid & 63;
  for (int t0 = 0; t0 < TT; t0 += 4) {
    const int t = t0 + tl;
    const float* qfr = qf + ((size_t)(b * TT + t) * HH + h) * FF;
    float zden = 1e-6f, av = 0.f;
#pragma unroll
    for (int f = 0; f < FF; ++f) {
      const float qv = qfr[f];
      zden += qv * Lks[f];
      av   += qv * Lkv[f * DH + d];
    }
    a[((size_t)(b * TT + t) * HH + h) * DH + d] = av / zden;
  }
}

// ---------------- x = LN(x [+ r]) * w + b ----------------
__global__ __launch_bounds__(256) void k_res_ln(float* __restrict__ x,
                                                const float* __restrict__ r,
                                                const float* __restrict__ w,
                                                const float* __restrict__ bsh,
                                                int addRes) {
  __shared__ float red[256], red2[256];
  const size_t row = blockIdx.x;
  float* xr = x + row * DD;
  const float* rr = r + row * DD;
  const int tid = threadIdx.x;
  float vals[3], s = 0.f, sq = 0.f;
#pragma unroll
  for (int i = 0; i < 3; ++i) {
    const int j = tid + i * 256;
    float v = xr[j];
    if (addRes) v += rr[j];
    vals[i] = v; s += v; sq += v * v;
  }
  red[tid] = s; red2[tid] = sq;
  __syncthreads();
  for (int off = 128; off > 0; off >>= 1) {
    if (tid < off) { red[tid] += red[tid + off]; red2[tid] += red2[tid + off]; }
    __syncthreads();
  }
  const float mean = red[0] * (1.f / DD);
  const float var  = red2[0] * (1.f / DD) - mean * mean;
  const float rs   = rsqrtf(var + 1e-5f);
#pragma unroll
  for (int i = 0; i < 3; ++i) {
    const int j = tid + i * 256;
    xr[j] = (vals[i] - mean) * rs * w[j] + bsh[j];
  }
}

// ---------------- masked mean pool over T ----------------
__global__ __launch_bounds__(256) void k_pool(const float* __restrict__ x,
                                              const int* __restrict__ mask,
                                              float* __restrict__ out) {
  const int b = blockIdx.x;
  const int tid = threadIdx.x;
  float cnt = 0.f, acc[3] = {0.f, 0.f, 0.f};
  for (int t = 0; t < TT; ++t) {
    const float mv = (float)mask[b * TT + t];
    cnt += mv;
    const float* row = x + ((size_t)b * TT + t) * DD;
#pragma unroll
    for (int i = 0; i < 3; ++i) acc[i] += row[tid + i * 256] * mv;
  }
  const float denom = fmaxf(cnt, 1e-9f);
#pragma unroll
  for (int i = 0; i < 3; ++i) out[(size_t)b * DD + tid + i * 256] = acc[i] / denom;
}

// ---------------- driver ----------------
extern "C" void kernel_launch(void* const* d_in, const int* in_sizes, int n_in,
                              void* d_out, int out_size, void* d_ws, size_t ws_size,
                              hipStream_t stream) {
  (void)in_sizes; (void)n_in; (void)out_size; (void)ws_size;
  const int*   ids   = (const int*)d_in[0];
  const int*   amask = (const int*)d_in[1];
  const float* temb  = (const float*)d_in[2];
  const float* Wq    = (const float*)d_in[3];
  const float* bq    = (const float*)d_in[4];
  const float* Wk    = (const float*)d_in[5];
  const float* bk    = (const float*)d_in[6];
  const float* Wv    = (const float*)d_in[7];
  const float* bv    = (const float*)d_in[8];
  const float* Wo    = (const float*)d_in[9];
  const float* bo    = (const float*)d_in[10];
  const float* omega = (const float*)d_in[11];
  const float* ln1w  = (const float*)d_in[12];
  const float* ln1b  = (const float*)d_in[13];
  const float* W1    = (const float*)d_in[14];
  const float* b1    = (const float*)d_in[15];
  const float* W2    = (const float*)d_in[16];
  const float* b2    = (const float*)d_in[17];
  const float* ln2w  = (const float*)d_in[18];
  const float* ln2b  = (const float*)d_in[19];
  const float* lnfw  = (const float*)d_in[20];
  const float* lnfb  = (const float*)d_in[21];

  const size_t MD = (size_t)MM * DD;
  const size_t MHF = (size_t)MM * HH * FF;
  float* ws   = (float*)d_ws;
  float* x    = ws;            // [M,D]
  float* bufq = x + MD;        // q  -> a
  float* bufk = bufq + MD;     // k  -> aout -> y2
  float* bufv = bufk + MD;     // v  -> y
  float* qf   = bufv + MD;     // [M,H,F]
  float* kf   = qf + MHF;      // [M,H,F]
  float* kv   = kf + MHF;      // [B,H,F,DH]
  float* ksum = kv + (size_t)BB * HH * FF * DH;  // [B,H,F]

  const dim3 gg(DD / TN, MM / TM);   // (6, 256)

  k_embed<<<MM, 256, 0, stream>>>(ids, temb, x);

  for (int i = 0; i < LL; ++i) {
    const size_t wo = (size_t)i * DD * DD;
    const size_t bo_ = (size_t)i * DD;
    // projections
    k_gemm<0><<<gg, 256, 0, stream>>>(x, Wq + wo, bq + bo_, bufq, MM, DD, DD);
    k_gemm<0><<<gg, 256, 0, stream>>>(x, Wk + wo, bk + bo_, bufk, MM, DD, DD);
    k_gemm<0><<<gg, 256, 0, stream>>>(x, Wv + wo, bv + bo_, bufv, MM, DD, DD);
    // RoPE + feature maps
    k_rope_feat<<<MM, 256, 0, stream>>>(bufq, bufk, omega + (size_t)i * DH * FF,
                                        amask, qf, kf);
    // linear attention
    k_kv_reduce<<<BB * HH, 256, 0, stream>>>(kf, bufv, kv, ksum);
    k_attn_out<<<BB * HH, 256, 0, stream>>>(qf, kv, ksum, bufq /* a */);
    // output projection + residual LN
    k_gemm<0><<<gg, 256, 0, stream>>>(bufq, Wo + wo, bo + bo_, bufk, MM, DD, DD);
    k_res_ln<<<MM, 256, 0, stream>>>(x, bufk, ln1w + bo_, ln1b + bo_, 1);
    // FFN (W1 + exact GELU fused, then W2) + residual LN
    k_gemm<1><<<gg, 256, 0, stream>>>(x, W1 + wo, b1 + bo_, bufv, MM, DD, DD);
    k_gemm<0><<<gg, 256, 0, stream>>>(bufv, W2 + wo, b2 + bo_, bufk, MM, DD, DD);
    k_res_ln<<<MM, 256, 0, stream>>>(x, bufk, ln2w + bo_, ln2b + bo_, 1);
  }

  k_res_ln<<<MM, 256, 0, stream>>>(x, x, lnfw, lnfb, 0);
  k_pool<<<BB, 256, 0, stream>>>(x, amask, (float*)d_out);
}
